// GNN_graphpred_73607149519515
// MI455X (gfx1250) — matrix operand order repaired and checked by reading, not compiled, not run on hardware
//
#include <hip/hip_runtime.h>
#include <hip/hip_bf16.h>

typedef __attribute__((ext_vector_type(16))) _Float16 v16h;
typedef __attribute__((ext_vector_type(8)))  float    v8f;

// ---------------------------------------------------------------------------
// Layer 0 edge kernel (5 input features): one thread per (edge, channel).
// msg = tanh(x[src]@Wx0 + (p[src]-p[dst])@Wp0 + ea@We0 + b0); sim[dst] += msg
// ---------------------------------------------------------------------------
__global__ __launch_bounds__(256) void edge_layer0(
    const float* __restrict__ x, const float* __restrict__ p,
    const float* __restrict__ ea, const int* __restrict__ src,
    const int* __restrict__ dst, const float* __restrict__ Wx0,
    const float* __restrict__ Wp0, const float* __restrict__ We0,
    const float* __restrict__ b0, float* __restrict__ sim, long nE)
{
    long idx = (long)blockIdx.x * blockDim.x + threadIdx.x;
    long e = idx >> 5;
    int  c = (int)(idx & 31);
    if (e >= nE) return;
    int s = src[e], d = dst[e];
    float acc = b0[c] + ea[e] * We0[c];
#pragma unroll
    for (int f = 0; f < 5; ++f)
        acc += x[(size_t)s * 5 + f] * Wx0[f * 32 + c];
#pragma unroll
    for (int j = 0; j < 3; ++j)
        acc += (p[(size_t)s * 3 + j] - p[(size_t)d * 3 + j]) * Wp0[j * 32 + c];
    atomicAdd(&sim[(size_t)d * 32 + c], tanhf(acc));
}

// ---------------------------------------------------------------------------
// Layers 1/2 edge kernel: WMMA f32_16x16x32_f16 (K == 32 exactly).
// One wave per 16-edge tile; A = h[src] tile (f32 -> f16), B = Wx column tiles
// kept in registers for the whole wave. Epilogue adds geometry terms, tanh,
// and scatters with fp32 L2 atomics.
// ---------------------------------------------------------------------------
__global__ __launch_bounds__(256) void edge_msg_wmma(
    const float* __restrict__ h, const float* __restrict__ p,
    const float* __restrict__ ea, const int* __restrict__ src,
    const int* __restrict__ dst, const float* __restrict__ Wx,
    const float* __restrict__ Wp, const float* __restrict__ We,
    const float* __restrict__ b, float* __restrict__ sim, long nE)
{
    const int lane = threadIdx.x & 31;
    const int wv   = threadIdx.x >> 5;
    const long tile = (long)blockIdx.x * 8 + wv;
    const long e0 = tile * 16;
    if (e0 >= nE) return;                 // tile-uniform: whole wave exits

    const int col = lane & 15;            // output column (N) / A row (M)
    const int hi  = lane >> 4;            // half-wave selector

    // ---- B tiles of Wx (32x32 row-major [k][n]), f16 WMMA layout:
    // lanes 0-15: col=lane, K=0..7 then 16..23 ; lanes 16-31: col=lane-16, K=8..15 then 24..31
    v16h B0, B1;
#pragma unroll
    for (int k = 0; k < 16; ++k) {
        int kk = (k & 8) * 2 + hi * 8 + (k & 7);
        B0[k] = (_Float16)Wx[kk * 32 + col];
        B1[k] = (_Float16)Wx[kk * 32 + col + 16];
    }

    // ---- A tile: 16 edges x 32 features of h[src], ISA 16-bit A layout:
    // lane<16: row=lane, K = 0..7 then 16..23 ; lane>=16: K = 8..15 then 24..31
    long eA = e0 + col; if (eA >= nE) eA = nE - 1;
    const int sA = src[eA];
    const float* hrow = h + (size_t)sA * 32 + hi * 8;
    v16h A;
#pragma unroll
    for (int k = 0; k < 8; ++k) {
        A[k]     = (_Float16)hrow[k];
        A[8 + k] = (_Float16)hrow[16 + k];
    }

    v8f acc0 = {};
    v8f acc1 = {};
    acc0 = __builtin_amdgcn_wmma_f32_16x16x32_f16(false, A, false, B0,
                                                  (short)0, acc0, false, false);
    acc1 = __builtin_amdgcn_wmma_f32_16x16x32_f16(false, A, false, B1,
                                                  (short)0, acc1, false, false);

    // column-invariant weights for this lane's two output columns
    const float wp00 = Wp[0 * 32 + col],      wp10 = Wp[1 * 32 + col],      wp20 = Wp[2 * 32 + col];
    const float wp01 = Wp[0 * 32 + col + 16], wp11 = Wp[1 * 32 + col + 16], wp21 = Wp[2 * 32 + col + 16];
    const float we0c = We[col], we1c = We[col + 16];
    const float bc0  = b[col],  bc1  = b[col + 16];

    // C layout: lane (col, hi) holds rows m = hi*8 + r, r = 0..7
#pragma unroll
    for (int r = 0; r < 8; ++r) {
        long er = e0 + hi * 8 + r;
        if (er < nE) {
            int s = src[er], d = dst[er];
            float prx = p[(size_t)s * 3 + 0] - p[(size_t)d * 3 + 0];
            float pry = p[(size_t)s * 3 + 1] - p[(size_t)d * 3 + 1];
            float prz = p[(size_t)s * 3 + 2] - p[(size_t)d * 3 + 2];
            float ev  = ea[er];
            float m0 = tanhf(acc0[r] + prx * wp00 + pry * wp10 + prz * wp20 + ev * we0c + bc0);
            float m1 = tanhf(acc1[r] + prx * wp01 + pry * wp11 + prz * wp21 + ev * we1c + bc1);
            atomicAdd(&sim[(size_t)d * 32 + col],      m0);
            atomicAdd(&sim[(size_t)d * 32 + col + 16], m1);
        }
    }
}

// ---------------------------------------------------------------------------
// Second aggregation: h'[dst] += sim[src]. One thread per (edge, 4-channel
// chunk): float4 gather from L2-resident sim, 4 fp32 atomics out.
// ---------------------------------------------------------------------------
__global__ __launch_bounds__(256) void aggregate(
    const float* __restrict__ sim, const int* __restrict__ src,
    const int* __restrict__ dst, float* __restrict__ hnext, long nE)
{
    long idx = (long)blockIdx.x * blockDim.x + threadIdx.x;
    long e = idx >> 3;
    int  q = (int)(idx & 7) * 4;
    if (e >= nE) return;
    int s = src[e], d = dst[e];
    const float4 v = *(const float4*)(sim + (size_t)s * 32 + q);
    float* o = hnext + (size_t)d * 32 + q;
    atomicAdd(o + 0, v.x);
    atomicAdd(o + 1, v.y);
    atomicAdd(o + 2, v.z);
    atomicAdd(o + 3, v.w);
}

// ---------------------------------------------------------------------------
// Mean-pool accumulation: gsum[batch[n]] += h[n]; gcnt[batch[n]] += 1
// ---------------------------------------------------------------------------
__global__ __launch_bounds__(256) void pool_accum(
    const float* __restrict__ h, const int* __restrict__ batch,
    float* __restrict__ gsum, float* __restrict__ gcnt, long nN)
{
    long idx = (long)blockIdx.x * blockDim.x + threadIdx.x;
    long n = idx >> 5;
    int  c = (int)(idx & 31);
    if (n >= nN) return;
    int g = batch[n];
    atomicAdd(&gsum[(size_t)g * 32 + c], h[(size_t)n * 32 + c]);
    if (c == 0) atomicAdd(&gcnt[g], 1.0f);
}

// ---------------------------------------------------------------------------
// Finalize: graph_rep = gsum / max(cnt,1); pred = graph_rep @ Wh + bh.
// One graph per wave32; pred via __shfl_xor wave reduction.
// ---------------------------------------------------------------------------
__global__ __launch_bounds__(256) void finalize(
    const float* __restrict__ gsum, const float* __restrict__ gcnt,
    const float* __restrict__ Wh, const float* __restrict__ bh,
    float* __restrict__ pred, float* __restrict__ rep, int G)
{
    int idx = blockIdx.x * blockDim.x + threadIdx.x;
    int g = idx >> 5;
    int c = idx & 31;
    if (g >= G) return;
    float cnt = gcnt[g];
    if (cnt < 1.0f) cnt = 1.0f;
    float r = gsum[(size_t)g * 32 + c] / cnt;
    rep[(size_t)g * 32 + c] = r;
    float t = r * Wh[c];
#pragma unroll
    for (int off = 16; off > 0; off >>= 1)
        t += __shfl_xor(t, off, 32);
    if (c == 0) pred[g] = t + bh[0];
}

// ---------------------------------------------------------------------------
extern "C" void kernel_launch(void* const* d_in, const int* in_sizes, int n_in,
                              void* d_out, int out_size, void* d_ws, size_t ws_size,
                              hipStream_t stream)
{
    const float* x   = (const float*)d_in[0];
    const float* p   = (const float*)d_in[1];
    const float* ea  = (const float*)d_in[2];
    const float* Wx0 = (const float*)d_in[3];
    const float* Wx1 = (const float*)d_in[4];
    const float* Wx2 = (const float*)d_in[5];
    const float* Wp0 = (const float*)d_in[6];
    const float* Wp1 = (const float*)d_in[7];
    const float* Wp2 = (const float*)d_in[8];
    const float* We0 = (const float*)d_in[9];
    const float* We1 = (const float*)d_in[10];
    const float* We2 = (const float*)d_in[11];
    const float* b0  = (const float*)d_in[12];
    const float* b1  = (const float*)d_in[13];
    const float* b2  = (const float*)d_in[14];
    const float* Wh  = (const float*)d_in[15];
    const float* bh  = (const float*)d_in[16];
    const int*   ei  = (const int*)d_in[17];
    const int*   bat = (const int*)d_in[18];

    const long nN = in_sizes[0] / 5;        // 100000
    const long nE = in_sizes[17] / 2;       // 3200000
    const int  G  = out_size / 33;          // 1024 (pred + 32*rep per graph)

    const int* src = ei;
    const int* dst = ei + nE;

    // workspace: sim[N*32], hbuf[N*32] (ping-pong via kill-then-zero), gsum, gcnt
    float* sim  = (float*)d_ws;
    float* hbuf = sim  + (size_t)nN * 32;
    float* gsum = hbuf + (size_t)nN * 32;
    float* gcnt = gsum + (size_t)G * 32;

    float* pred = (float*)d_out;            // [G]
    float* rep  = pred + G;                 // [G,32]

    const size_t nkBytes = (size_t)nN * 32 * sizeof(float);

    const int blk = 256;
    const int gEdgeC  = (int)((nE * 32 + blk - 1) / blk);   // (edge, chan) threads
    const long tiles  = (nE + 15) / 16;
    const int gWmma   = (int)((tiles + 7) / 8);             // 8 waves / block
    const int gAgg    = (int)((nE * 8 + blk - 1) / blk);    // (edge, chan/4)
    const int gPool   = (int)((nN * 32 + blk - 1) / blk);
    const int gFin    = (G * 32 + blk - 1) / blk;

    // ---- layer 0
    hipMemsetAsync(sim, 0, nkBytes, stream);
    edge_layer0<<<gEdgeC, blk, 0, stream>>>(x, p, ea, src, dst, Wx0, Wp0, We0, b0, sim, nE);
    hipMemsetAsync(hbuf, 0, nkBytes, stream);
    aggregate<<<gAgg, blk, 0, stream>>>(sim, src, dst, hbuf, nE);

    // ---- layer 1 (WMMA)
    hipMemsetAsync(sim, 0, nkBytes, stream);
    edge_msg_wmma<<<gWmma, blk, 0, stream>>>(hbuf, p, ea, src, dst, Wx1, Wp1, We1, b1, sim, nE);
    hipMemsetAsync(hbuf, 0, nkBytes, stream);
    aggregate<<<gAgg, blk, 0, stream>>>(sim, src, dst, hbuf, nE);

    // ---- layer 2 (WMMA)
    hipMemsetAsync(sim, 0, nkBytes, stream);
    edge_msg_wmma<<<gWmma, blk, 0, stream>>>(hbuf, p, ea, src, dst, Wx2, Wp2, We2, b2, sim, nE);
    hipMemsetAsync(hbuf, 0, nkBytes, stream);
    aggregate<<<gAgg, blk, 0, stream>>>(sim, src, dst, hbuf, nE);

    // ---- pooling + head
    hipMemsetAsync(gsum, 0, (size_t)G * 33 * sizeof(float), stream); // gsum + gcnt contiguous
    pool_accum<<<gPool, blk, 0, stream>>>(hbuf, bat, gsum, gcnt, nN);
    finalize<<<gFin, blk, 0, stream>>>(gsum, gcnt, Wh, bh, pred, rep, G);
}
